// ResonanceLayer_9414568313181
// MI455X (gfx1250) — matrix operand order repaired and checked by reading, not compiled, not run on hardware
//
#include <hip/hip_runtime.h>
#include <hip/hip_bf16.h>

// ---------------------------------------------------------------------------
// ResonanceLayer for MI455X (gfx1250, wave32, WMMA).
//   K0: one-time w_freq fp32 -> bf16 image in the LDS staging layout
//   K1: fused bf16-WMMA projection + row norms + pairwise-cosine partials
//       (software-pipelined staging, ping-pong B fragments, copy-only W path)
//   K2: deterministic reduction -> 8x8 resonance matrix (0.1/B folded in)
//   K3: streaming enhance pass (reads emb once, writes out once)
// ---------------------------------------------------------------------------

#define N_MOD 8
#define BATCH 32768
#define HID   512
#define TB    16                   // batch rows per block in K1
#define K1_THREADS 1024            // 32 waves: 8 row-tiles x 4 col-groups
#define K1_BLOCKS  (BATCH / TB)    // 2048
#define NPAIR 28                   // 8*7/2 off-diagonal pairs
#define WIMG_U32 (16 * 512 * 20)   // 16 chunks x 512 rows x 20 u32 (40 halves)

typedef __attribute__((ext_vector_type(16))) __bf16 v16bf;
typedef __attribute__((ext_vector_type(8)))  float  v8f;

union FragU { uint4 q[2]; v16bf v; };

__device__ __forceinline__ unsigned pack_bf2(float a, float b) {
    unsigned ua = __float_as_uint(a); ua += 0x7FFFu + ((ua >> 16) & 1u);  // RNE
    unsigned ub = __float_as_uint(b); ub += 0x7FFFu + ((ub >> 16) & 1u);
    return (ua >> 16) | (ub & 0xFFFF0000u);
}

// LDS partition (bytes):
//   proj  : 512 cols * 68 u32 (128 rows packed 2x bf16, pad 64->68)  = 139264
//   Wst   : 512 rows * 20 u32 (32 bf16 padded to 40 halves)          =  40960
//   Ast   : 128 rows * 20 u32                                        =  10240
//   bias 512 | sumsq 4*128 | invn 128 | red 28*32                    =   8192
#define PROJ_U32 (512 * 68)
#define WST_U32  (512 * 20)
#define AST_U32  (128 * 20)
#define SMEM1_BYTES ((PROJ_U32 + WST_U32 + AST_U32) * 4 + (512 + 512 + 128 + 896) * 4)

// ---------------- K0: w_freq -> bf16 staging image --------------------------
// wimg[kc*10240 + e*20 + j] : u32 j<16 holds bf16{w[e, kc*32+2j], w[e, kc*32+2j+1]},
// j in 16..19 is the LDS bank-conflict padding (zero).
__global__ __launch_bounds__(256)
void wfreq_convert_kernel(const float* __restrict__ wfr, unsigned* __restrict__ wimg) {
    int idx = blockIdx.x * 256 + threadIdx.x;   // 0..163839
    int kc = idx / 10240;
    int r  = idx - kc * 10240;
    int e  = r / 20;
    int j  = r - e * 20;
    unsigned v = 0u;
    if (j < 16) {
        const float* src = wfr + (size_t)e * HID + kc * 32 + j * 2;
        v = pack_bf2(src[0], src[1]);
    }
    wimg[idx] = v;
}

// Per-thread staging buffer for one k-chunk (1 A float4 + 5 W uint2 copies).
struct Stage { float4 a; uint2 w[5]; };

__device__ __forceinline__ void load_stage(Stage& st, const float4* __restrict__ emb4,
                                           const uint2* __restrict__ wimg2,
                                           int tid, int b0, int kc) {
    {   // A: 128 rows x 8 float4 = 1024 float4 -> one per thread (fp32, convert later)
        int row = tid >> 3, k4 = tid & 7;
        int n = row >> 4, rb = row & 15;
        st.a = emb4[((size_t)n * BATCH + b0 + rb) * (HID / 4) + kc * 8 + k4];
    }
    const uint2* src = wimg2 + (size_t)kc * 5120;   // pre-converted bf16 image: pure copy
#pragma unroll
    for (int s = 0; s < 5; ++s) st.w[s] = src[tid + s * K1_THREADS];
}

__device__ __forceinline__ void store_stage(const Stage& st, unsigned* Ast, unsigned* Wst,
                                            int tid) {
    {
        int row = tid >> 3, k4 = tid & 7;
        uint2 p; p.x = pack_bf2(st.a.x, st.a.y); p.y = pack_bf2(st.a.z, st.a.w);
        *(uint2*)&Ast[row * 20 + k4 * 2] = p;
    }
#pragma unroll
    for (int s = 0; s < 5; ++s)
        *(uint2*)&Wst[(tid + s * K1_THREADS) * 2] = st.w[s];
}

// One k-chunk of WMMAs for this wave: 8 column tiles, ping-pong B fragments so
// the ds_load_b128 pair for tile ct+1 overlaps the WMMA for tile ct.
__device__ __forceinline__ void wmma_chunk(v8f* acc, const unsigned* Ast, const unsigned* Wst,
                                           int rt, int cg, int lan, int half) {
    FragU af;
    {   // A fragment: lane<16 -> K[0..7],[16..23]; lane>=16 -> K[8..15],[24..31]
        int row = rt * 16 + lan;
        int base = row * 20 + half * 4;
        af.q[0] = *(const uint4*)&Ast[base];
        af.q[1] = *(const uint4*)&Ast[base + 8];
    }
    auto loadB = [&](FragU& fb, int ct) {  // B: lane<16 -> col=lane K[0..7],[16..23]; else K[8..15],[24..31]
        int e = cg * 128 + ct * 16 + lan;
        int base = e * 20 + half * 4;
        fb.q[0] = *(const uint4*)&Wst[base];
        fb.q[1] = *(const uint4*)&Wst[base + 8];
    };
    FragU fb0, fb1;
    loadB(fb0, 0);
#pragma unroll
    for (int ct = 0; ct < 8; ct += 2) {
        loadB(fb1, ct + 1);
        acc[ct] = __builtin_amdgcn_wmma_f32_16x16x32_bf16(
            false, af.v, false, fb0.v, (short)0, acc[ct], false, false);
        if (ct + 2 < 8) loadB(fb0, ct + 2);
        acc[ct + 1] = __builtin_amdgcn_wmma_f32_16x16x32_bf16(
            false, af.v, false, fb1.v, (short)0, acc[ct + 1], false, false);
    }
}

__global__ __launch_bounds__(K1_THREADS, 1)
void resonance_gemm_gram_kernel(const float* __restrict__ emb,
                                const unsigned* __restrict__ wimg,
                                const float* __restrict__ bfr,
                                float* __restrict__ partials) {
    extern __shared__ __align__(16) char smem[];
    unsigned* proj  = (unsigned*)smem;
    unsigned* Wst   = proj + PROJ_U32;
    unsigned* Ast   = Wst + WST_U32;
    float*    bias  = (float*)(Ast + AST_U32);
    float*    sumsq = bias + 512;       // [4 col-groups][128 rows]
    float*    invn  = sumsq + 512;
    float*    red   = invn + 128;       // [28 pairs][32]

    const int tid  = threadIdx.x;
    const int lane = tid & 31;
    const int wave = tid >> 5;
    const int rt   = wave & 7;    // row tile (16 rows)
    const int cg   = wave >> 3;   // column group (128 cols)
    const int lan  = lane & 15;
    const int half = lane >> 4;
    const int b0   = blockIdx.x * TB;

    const float4* emb4  = (const float4*)emb;
    const uint2*  wimg2 = (const uint2*)wimg;

    if (tid < 512) bias[tid] = bfr[tid];

    v8f acc[8];
#pragma unroll
    for (int ct = 0; ct < 8; ++ct)
#pragma unroll
        for (int v = 0; v < 8; ++v) acc[ct][v] = 0.0f;

    // ---------------- K loop: 16 chunks of 32, pipelined 2 deep ----------------
    Stage s0, s1;
    load_stage(s0, emb4, wimg2, tid, b0, 0);
    for (int kc = 0; kc < 16; kc += 2) {
        store_stage(s0, Ast, Wst, tid);
        __syncthreads();
        load_stage(s1, emb4, wimg2, tid, b0, kc + 1);   // overlaps WMMAs below
        wmma_chunk(acc, Ast, Wst, rt, cg, lan, half);
        __syncthreads();

        store_stage(s1, Ast, Wst, tid);
        __syncthreads();
        if (kc + 2 < 16) load_stage(s0, emb4, wimg2, tid, b0, kc + 2);
        wmma_chunk(acc, Ast, Wst, rt, cg, lan, half);
        __syncthreads();
    }

    // ---------------- bias + row sum-of-squares (fp32, pre-bf16) ----------------
    float sq[8] = {0, 0, 0, 0, 0, 0, 0, 0};
#pragma unroll
    for (int ct = 0; ct < 8; ++ct) {
        float bv = bias[cg * 128 + ct * 16 + lan];
#pragma unroll
        for (int v = 0; v < 8; ++v) {
            float x = acc[ct][v] + bv;
            acc[ct][v] = x;
            sq[v] = fmaf(x, x, sq[v]);
        }
    }
#pragma unroll
    for (int v = 0; v < 8; ++v)
#pragma unroll
        for (int m = 1; m < 16; m <<= 1) sq[v] += __shfl_xor(sq[v], m, 32);
    if (lan == 0) {  // unique (cg,row) slots -> no atomics, deterministic
        int rb_ = rt * 16 + half * 8;
#pragma unroll
        for (int v = 0; v < 8; ++v) sumsq[cg * 128 + rb_ + v] = sq[v];
    }

    // store proj as packed bf16, transposed [col][rowpair]
#pragma unroll
    for (int ct = 0; ct < 8; ++ct) {
        int c = cg * 128 + ct * 16 + lan;
        uint4 q;
        q.x = pack_bf2(acc[ct][0], acc[ct][1]);
        q.y = pack_bf2(acc[ct][2], acc[ct][3]);
        q.z = pack_bf2(acc[ct][4], acc[ct][5]);
        q.w = pack_bf2(acc[ct][6], acc[ct][7]);
        *(uint4*)&proj[c * 68 + rt * 8 + half * 4] = q;
    }
    __syncthreads();

    if (tid < 128) {   // fixed-order 4-way sum -> deterministic
        float s = sumsq[tid] + sumsq[128 + tid];
        s += sumsq[256 + tid] + sumsq[384 + tid];
        invn[tid] = 1.0f / fmaxf(sqrtf(s), 1e-8f);
    }
    __syncthreads();

    // ---------------- pairwise gram partials (2-way d split) ----------------
    if (tid < NPAIR * 32) {
        int p = tid >> 5, sub = tid & 31;
        int b = sub & 15, dh = sub >> 4;
        int pi = p, i = 0, span = 7;
        while (pi >= span) { pi -= span; ++i; --span; }
        int j = i + 1 + pi;
        const int sel = b & 1;
        const int dwi = i * 8 + (b >> 1);
        const int dwj = j * 8 + (b >> 1);
        float dot = 0.0f;
        const int dbase = dh * 256;
        for (int d = 0; d < 256; ++d) {
            unsigned wi = proj[(dbase + d) * 68 + dwi];
            unsigned wj = proj[(dbase + d) * 68 + dwj];
            float fi = __uint_as_float(sel ? (wi & 0xFFFF0000u) : (wi << 16));
            float fj = __uint_as_float(sel ? (wj & 0xFFFF0000u) : (wj << 16));
            dot = fmaf(fi, fj, dot);
        }
        red[tid] = dot * invn[i * 16 + b] * invn[j * 16 + b];
    }
    __syncthreads();
    if (tid < NPAIR) {
        float s = 0.0f;
#pragma unroll
        for (int q = 0; q < 32; ++q) s += red[tid * 32 + q];
        partials[(size_t)blockIdx.x * NPAIR + tid] = s;
    }
}

__global__ void resonance_reduce_kernel(const float* __restrict__ partials,
                                        float* __restrict__ res) {
    int t = threadIdx.x;
    if (t < NPAIR) {
        float s = 0.0f;
        for (int blk = 0; blk < K1_BLOCKS; ++blk) s += partials[(size_t)blk * NPAIR + t];
        int pi = t, i = 0, span = 7;
        while (pi >= span) { pi -= span; ++i; --span; }
        int j = i + 1 + pi;
        float v = 0.1f * s / (float)BATCH;   // RESONANCE_STRENGTH * sim/B
        res[i * 8 + j] = v;
        res[j * 8 + i] = v;
    }
    if (t < 8) res[t * 9] = 0.0f;
}

__global__ __launch_bounds__(256)
void resonance_enhance_kernel(const float4* __restrict__ emb,
                              const float4* __restrict__ ecow,
                              const float* __restrict__ res,
                              float4* __restrict__ out) {
    const size_t slice = (size_t)BATCH * HID / 4;
    const size_t f = (size_t)blockIdx.x * 256 + threadIdx.x;
    float r[64];
#pragma unroll
    for (int q = 0; q < 64; ++q) r[q] = res[q];   // uniform -> scalar loads
    float4 e[8];
#pragma unroll
    for (int i = 0; i < 8; ++i) e[i] = emb[slice * i + f];
    float4 w = ecow[f & (HID / 4 - 1)];
#pragma unroll
    for (int i = 0; i < 8; ++i) {
        float ax = 0.f, ay = 0.f, az = 0.f, aw = 0.f;
#pragma unroll
        for (int j = 0; j < 8; ++j) {
            float c = r[i * 8 + j];
            ax = fmaf(c, e[j].x, ax);
            ay = fmaf(c, e[j].y, ay);
            az = fmaf(c, e[j].z, az);
            aw = fmaf(c, e[j].w, aw);
        }
        float4 o;
        o.x = e[i].x + ax * w.x;
        o.y = e[i].y + ay * w.y;
        o.z = e[i].z + az * w.z;
        o.w = e[i].w + aw * w.w;
        out[slice * i + f] = o;
    }
}

extern "C" void kernel_launch(void* const* d_in, const int* in_sizes, int n_in,
                              void* d_out, int out_size, void* d_ws, size_t ws_size,
                              hipStream_t stream) {
    const float* emb = (const float*)d_in[0];   // [8, 32768, 512]
    const float* wfr = (const float*)d_in[1];   // [512, 512]
    const float* bfr = (const float*)d_in[2];   // [512]
    const float* eco = (const float*)d_in[3];   // [1, 512]
    float* out = (float*)d_out;

    // workspace layout (floats): res[64] | partials[2048*28] | w bf16 image[163840 u32]
    float*    res      = (float*)d_ws;
    float*    partials = res + 64;
    unsigned* wimg     = (unsigned*)(partials + (size_t)K1_BLOCKS * NPAIR);

    (void)hipFuncSetAttribute((const void*)resonance_gemm_gram_kernel,
                              hipFuncAttributeMaxDynamicSharedMemorySize,
                              (int)SMEM1_BYTES);

    wfreq_convert_kernel<<<WIMG_U32 / 256, 256, 0, stream>>>(wfr, wimg);
    resonance_gemm_gram_kernel<<<K1_BLOCKS, K1_THREADS, SMEM1_BYTES, stream>>>(
        emb, wimg, bfr, partials);
    resonance_reduce_kernel<<<1, 32, 0, stream>>>(partials, res);
    resonance_enhance_kernel<<<(BATCH * HID / 4) / 256, 256, 0, stream>>>(
        (const float4*)emb, (const float4*)eco, res, (float4*)out);
}
